// BotRGCN_40699110097106
// MI455X (gfx1250) — matrix operand order repaired and checked by reading, not compiled, not run on hardware
//
#include <hip/hip_runtime.h>

// BotRGCN forward for MI455X (gfx1250, wave32).
// - All dense matmuls via v_wmma_f32_16x16x32_f16 (one wave owns a 16-node tile).
// - Weight B-fragments are pre-packed once per launch into WMMA register order
//   (f16), so hot loops do pure b128 vector loads + WMMA (no per-tile strided
//   scalar weight loads / f32->f16 converts).
// - Edge aggregation: L2-resident gather + global f32 atomics; degree counts
//   computed once and reused for both RGCN layers.
// Requires N % 16 == 0 (N = 100000) and ws_size >= (194*N*4 + 64KB) bytes.

typedef __attribute__((ext_vector_type(16))) _Float16 v16h;
typedef __attribute__((ext_vector_type(8)))  float    v8f;

#define WMMA_F16(a, b, c) \
  __builtin_amdgcn_wmma_f32_16x16x32_f16(false, (a), false, (b), (short)0, (c), false, false)

__device__ __forceinline__ float lrelu(float v) { return v > 0.0f ? v : 0.01f * v; }

// ---- WMMA fragment helpers (ISA 7.12.2 layouts, wave32) -------------------
// A: 16x32 f16. lane&15 = row; lanes 0-15 hold K={0..7,16..23}, lanes 16-31
// hold K={8..15,24..31}; v16h element e -> K = (e<8?0:16) + 8*(lane>>4) + (e&7).
__device__ __forceinline__ v16h load_A_f32(const float* __restrict__ X, int ld,
                                           int row0, int koff) {
  const int lane = threadIdx.x & 31;
  const int half = lane >> 4;
  const int row  = row0 + (lane & 15);
  const float* p = X + (size_t)row * ld + koff + half * 8;
  v16h a;
#pragma unroll
  for (int e = 0; e < 8; ++e) a[e] = (_Float16)p[e];
#pragma unroll
  for (int e = 0; e < 8; ++e) a[8 + e] = (_Float16)p[16 + e];
  return a;
}

// B: 32x16 f16 from row-major W[K x ldw]. lane&15 = col; lanes 0-15 hold
// K={0..7,16..23}, lanes 16-31 hold K={8..15,24..31}; element e -> K = (e<8?0:16) + 8*(lane>>4) + (e&7).
// Columns >= ncols are zero-padded (8-wide and 2-wide outputs).
__device__ __forceinline__ v16h make_B_f32(const float* __restrict__ W, int ldw,
                                           int ncols, int koff, int col0) {
  const int lane = threadIdx.x & 31;
  const int col  = col0 + (lane & 15);
  const int kb   = koff + ((lane >> 4) << 3);
  v16h b;
  if (col < ncols) {
#pragma unroll
    for (int e = 0; e < 16; ++e) b[e] = (_Float16)W[(size_t)(kb + e + (e & 8)) * ldw + col];
  } else {
#pragma unroll
    for (int e = 0; e < 16; ++e) b[e] = (_Float16)0.0f;
  }
  return b;
}

// Pre-packed fragment fetch: one aligned 32B vector load per lane.
__device__ __forceinline__ v16h load_frag(const _Float16* __restrict__ pack, int job) {
  return *(const v16h*)(pack + (size_t)job * 512 + (threadIdx.x & 31) * 16);
}

// Pack-buffer job indices (each job = 512 halfs = one wave-wide B fragment)
#define F_DES  0    // jobs 0..23   : W_des  K-chunks ko = j*32
#define F_TW   24   // jobs 24..47  : W_tweet K-chunks
#define F_IN   48   // 48: W_in cols 0-15, 49: cols 16-31
#define F_REL0 50   // 50,51: W_rel[0]
#define F_REL1 52   // 52,53: W_rel[1]
#define F_ROOT 54   // 54,55: W_root
#define F_O1   56   // 56,57: W_out1
#define F_O2   58   // 58: W_out2 (2 cols, zero-padded)
#define N_JOBS 59

// ---- prepack: convert all weight matrices into WMMA B-fragment images -----
__global__ __launch_bounds__(32) void prepack_kernel(
    const float* __restrict__ W_des, const float* __restrict__ W_tw,
    const float* __restrict__ W_in,  const float* __restrict__ W_rel,
    const float* __restrict__ W_root, const float* __restrict__ W_o1,
    const float* __restrict__ W_o2, _Float16* __restrict__ pack) {
  const int job = blockIdx.x;
  v16h b;
  if      (job < 24)  b = make_B_f32(W_des, 8, 8, job * 32, 0);
  else if (job < 48)  b = make_B_f32(W_tw, 8, 8, (job - 24) * 32, 0);
  else if (job == 48) b = make_B_f32(W_in, 32, 32, 0, 0);
  else if (job == 49) b = make_B_f32(W_in, 32, 32, 0, 16);
  else if (job == 50) b = make_B_f32(W_rel, 32, 32, 0, 0);
  else if (job == 51) b = make_B_f32(W_rel, 32, 32, 0, 16);
  else if (job == 52) b = make_B_f32(W_rel + 1024, 32, 32, 0, 0);
  else if (job == 53) b = make_B_f32(W_rel + 1024, 32, 32, 0, 16);
  else if (job == 54) b = make_B_f32(W_root, 32, 32, 0, 0);
  else if (job == 55) b = make_B_f32(W_root, 32, 32, 0, 16);
  else if (job == 56) b = make_B_f32(W_o1, 32, 32, 0, 0);
  else if (job == 57) b = make_B_f32(W_o1, 32, 32, 0, 16);
  else                b = make_B_f32(W_o2, 2, 2, 0, 0);
  *(v16h*)(pack + (size_t)job * 512 + (threadIdx.x & 31) * 16) = b;
}

// ---- zero scratch ----------------------------------------------------------
__global__ void zero_kernel(float* __restrict__ p, int n) {
  int i = blockIdx.x * blockDim.x + threadIdx.x;
  if (i < n) p[i] = 0.0f;
}

// ---- encoder: x = lrelu(cat(d,t,n,c) @ W_in + b_in) ------------------------
__global__ __launch_bounds__(32) void encoder_kernel(
    const float* __restrict__ des,  const float* __restrict__ tweet,
    const float* __restrict__ nump, const float* __restrict__ catp,
    const float* __restrict__ b_des, const float* __restrict__ b_tw,
    const float* __restrict__ W_num, const float* __restrict__ b_num,
    const float* __restrict__ W_cat, const float* __restrict__ b_cat,
    const float* __restrict__ b_in, const _Float16* __restrict__ pack,
    float* __restrict__ x) {
  __shared__ float xs[16][33];            // 16-node tile of x (pre-W_in), padded
  const int row0 = blockIdx.x * 16;
  const int lane = threadIdx.x;
  const int col  = lane & 15;
  const int rb   = (lane >> 4) * 8;       // C-frag row base (M = rb + vgpr)
  const size_t arow = (size_t)(row0 + col) * 768 + (lane >> 4) * 8;

  v8f cd = {}, ct = {};
#pragma unroll 1
  for (int ko = 0; ko < 768; ko += 32) {  // K=768 in 24 WMMA steps per stream
    if (ko + 32 < 768) {                  // speculative prefetch of next chunk
      __builtin_prefetch(des + arow + ko + 32, 0, 3);
      __builtin_prefetch(tweet + arow + ko + 32, 0, 3);
    }
    v16h ad = load_A_f32(des, 768, row0, ko);
    v16h bd = load_frag(pack, F_DES + (ko >> 5));
    cd = WMMA_F16(ad, bd, cd);
    v16h at = load_A_f32(tweet, 768, row0, ko);
    v16h bt = load_frag(pack, F_TW + (ko >> 5));
    ct = WMMA_F16(at, bt, ct);
  }
  if (col < 8) {                          // cols 0-7 = des, 8-15 = tweet
#pragma unroll
    for (int j = 0; j < 8; ++j) {
      xs[rb + j][col]     = lrelu(cd[j] + b_des[col]);
      xs[rb + j][8 + col] = lrelu(ct[j] + b_tw[col]);
    }
  }
  {                                       // num (K=5) / cat (K=3): scalar VALU
    const int c8 = lane & 7;
    const int r4 = (lane >> 3) * 4;
#pragma unroll
    for (int i = 0; i < 4; ++i) {
      const int r = r4 + i;
      const size_t g = (size_t)(row0 + r);
      float an = b_num[c8];
#pragma unroll
      for (int k = 0; k < 5; ++k) an += nump[g * 5 + k] * W_num[k * 8 + c8];
      xs[r][16 + c8] = lrelu(an);
      float ac = b_cat[c8];
#pragma unroll
      for (int k = 0; k < 3; ++k) ac += catp[g * 3 + k] * W_cat[k * 8 + c8];
      xs[r][24 + c8] = lrelu(ac);
    }
  }
  __syncthreads();
  // x = lrelu(xs @ W_in + b_in): one A-frag from LDS, two pre-packed B halves
  v16h a  = load_A_f32(&xs[0][0], 33, 0, 0);
  v16h b0 = load_frag(pack, F_IN);
  v16h b1 = load_frag(pack, F_IN + 1);
  v8f c0 = {}, c1 = {};
  c0 = WMMA_F16(a, b0, c0);
  c1 = WMMA_F16(a, b1, c1);
#pragma unroll
  for (int j = 0; j < 8; ++j) {
    const size_t r = (size_t)(row0 + rb + j);
    x[r * 32 + col]      = lrelu(c0[j] + b_in[col]);
    x[r * 32 + 16 + col] = lrelu(c1[j] + b_in[16 + col]);
  }
}

// ---- dense part of RGCN layer: h0 = x@Wrel0, h1 = x@Wrel1, root = x@Wroot+b -
__global__ __launch_bounds__(32) void transform3_kernel(
    const float* __restrict__ x, const _Float16* __restrict__ pack,
    const float* __restrict__ b_rg, float* __restrict__ h01,
    float* __restrict__ root, int N) {
  const int row0 = blockIdx.x * 16;
  const int lane = threadIdx.x;
  const int col  = lane & 15;
  const int rb   = (lane >> 4) * 8;
  v16h a = load_A_f32(x, 32, row0, 0);

  float* outs[3] = {h01, h01 + (size_t)N * 32, root};
#pragma unroll
  for (int m = 0; m < 3; ++m) {
    v16h b0 = load_frag(pack, F_REL0 + 2 * m);
    v16h b1 = load_frag(pack, F_REL0 + 2 * m + 1);
    v8f c0 = {}, c1 = {};
    c0 = WMMA_F16(a, b0, c0);
    c1 = WMMA_F16(a, b1, c1);
    const float bl = (m == 2) ? b_rg[col] : 0.0f;
    const float bh = (m == 2) ? b_rg[16 + col] : 0.0f;
#pragma unroll
    for (int j = 0; j < 8; ++j) {
      const size_t r = (size_t)(row0 + rb + j);
      outs[m][r * 32 + col]      = c0[j] + bl;
      outs[m][r * 32 + 16 + col] = c1[j] + bh;
    }
  }
}

// ---- edge scatter: sums[rel*N+dst,:] += h[rel][src,:] (+ cnt on first layer)
__global__ void edge_kernel(const int* __restrict__ edge_index,
                            const int* __restrict__ edge_type,
                            const float* __restrict__ h01,
                            float* __restrict__ sums, float* __restrict__ cnt,
                            int N, unsigned E, int addcnt) {
  const unsigned tid = blockIdx.x * blockDim.x + threadIdx.x;
  const unsigned e = tid >> 3;          // 8 threads per edge, 4 floats each
  if (e >= E) return;
  const int part = tid & 7;
  const int r = edge_type[e];
  const int s = edge_index[e];
  const int d = edge_index[(size_t)E + e];
  const float4 v = *(const float4*)(h01 + (((size_t)r * N + s) << 5) + part * 4);
  float* dp = sums + (((size_t)r * N + d) << 5) + part * 4;
  __hip_atomic_fetch_add(dp + 0, v.x, __ATOMIC_RELAXED, __HIP_MEMORY_SCOPE_AGENT);
  __hip_atomic_fetch_add(dp + 1, v.y, __ATOMIC_RELAXED, __HIP_MEMORY_SCOPE_AGENT);
  __hip_atomic_fetch_add(dp + 2, v.z, __ATOMIC_RELAXED, __HIP_MEMORY_SCOPE_AGENT);
  __hip_atomic_fetch_add(dp + 3, v.w, __ATOMIC_RELAXED, __HIP_MEMORY_SCOPE_AGENT);
  if (addcnt && part == 0)
    __hip_atomic_fetch_add(cnt + (size_t)r * N + d, 1.0f,
                           __ATOMIC_RELAXED, __HIP_MEMORY_SCOPE_AGENT);
}

// ---- combine: x = mean_rel0 + mean_rel1 + root -----------------------------
__global__ void combine_kernel(const float* __restrict__ sums,
                               const float* __restrict__ cnt,
                               const float* __restrict__ root,
                               float* __restrict__ x, int N) {
  const int i = blockIdx.x * blockDim.x + threadIdx.x;
  if (i >= N * 32) return;
  const int row = i >> 5;
  const float m0 = sums[i] / fmaxf(cnt[row], 1.0f);
  const float m1 = sums[(size_t)N * 32 + i] / fmaxf(cnt[(size_t)N + row], 1.0f);
  x[i] = m0 + m1 + root[i];
}

// ---- head: out = lrelu(x@W_out1+b1) @ W_out2 + b2 --------------------------
__global__ __launch_bounds__(32) void head_kernel(
    const float* __restrict__ x, const _Float16* __restrict__ pack,
    const float* __restrict__ b1, const float* __restrict__ b2,
    float* __restrict__ out) {
  __shared__ float ys[16][33];
  const int row0 = blockIdx.x * 16;
  const int lane = threadIdx.x;
  const int col  = lane & 15;
  const int rb   = (lane >> 4) * 8;
  v16h a  = load_A_f32(x, 32, row0, 0);
  v16h f0 = load_frag(pack, F_O1);
  v16h f1 = load_frag(pack, F_O1 + 1);
  v8f c0 = {}, c1 = {};
  c0 = WMMA_F16(a, f0, c0);
  c1 = WMMA_F16(a, f1, c1);
#pragma unroll
  for (int j = 0; j < 8; ++j) {
    ys[rb + j][col]      = lrelu(c0[j] + b1[col]);
    ys[rb + j][16 + col] = lrelu(c1[j] + b1[16 + col]);
  }
  __syncthreads();
  v16h a2 = load_A_f32(&ys[0][0], 33, 0, 0);
  v16h f2 = load_frag(pack, F_O2);        // 2 real cols, 14 zero-padded
  v8f c2 = {};
  c2 = WMMA_F16(a2, f2, c2);
  if (col < 2) {
#pragma unroll
    for (int j = 0; j < 8; ++j)
      out[(size_t)(row0 + rb + j) * 2 + col] = c2[j] + b2[col];
  }
}

// ---------------------------------------------------------------------------
extern "C" void kernel_launch(void* const* d_in, const int* in_sizes, int n_in,
                              void* d_out, int out_size, void* d_ws, size_t ws_size,
                              hipStream_t stream) {
  (void)n_in; (void)out_size; (void)ws_size;
  const float* des   = (const float*)d_in[0];
  const float* tweet = (const float*)d_in[1];
  const float* nump  = (const float*)d_in[2];
  const float* catp  = (const float*)d_in[3];
  const int*   eidx  = (const int*)d_in[4];
  const int*   etyp  = (const int*)d_in[5];
  const float* W_des = (const float*)d_in[6];  const float* b_des = (const float*)d_in[7];
  const float* W_tw  = (const float*)d_in[8];  const float* b_tw  = (const float*)d_in[9];
  const float* W_num = (const float*)d_in[10]; const float* b_num = (const float*)d_in[11];
  const float* W_cat = (const float*)d_in[12]; const float* b_cat = (const float*)d_in[13];
  const float* W_in  = (const float*)d_in[14]; const float* b_in  = (const float*)d_in[15];
  const float* W_rel = (const float*)d_in[16];
  const float* W_rt  = (const float*)d_in[17]; const float* b_rg  = (const float*)d_in[18];
  const float* W_o1  = (const float*)d_in[19]; const float* b_o1  = (const float*)d_in[20];
  const float* W_o2  = (const float*)d_in[21]; const float* b_o2  = (const float*)d_in[22];

  const int N = in_sizes[0] / 768;       // 100000 (multiple of 16)
  const unsigned E = (unsigned)in_sizes[5];

  // Workspace layout (floats): x | h0 | h1 | root | sums[2N*32] | cnt[2N] | pack
  float* ws   = (float*)d_ws;
  float* x    = ws;
  float* h01  = ws + (size_t)32 * N;
  float* root = ws + (size_t)96 * N;
  float* sums = ws + (size_t)128 * N;
  float* cnt  = ws + (size_t)192 * N;
  _Float16* pack = (_Float16*)(ws + (size_t)194 * N);  // N_JOBS*512 halfs (~60KB)

  const int tiles = N / 16;              // 6250 one-wave workgroups

  prepack_kernel<<<N_JOBS, 32, 0, stream>>>(W_des, W_tw, W_in, W_rel, W_rt,
                                            W_o1, W_o2, pack);
  {                                      // zero sums+cnt every launch
    const int n = 66 * N;
    zero_kernel<<<(n + 255) / 256, 256, 0, stream>>>(sums, n);
  }

  encoder_kernel<<<tiles, 32, 0, stream>>>(des, tweet, nump, catp,
                                           b_des, b_tw, W_num, b_num,
                                           W_cat, b_cat, b_in, pack, x);

  const unsigned egrid = (E * 8u + 255u) / 256u;
  for (int layer = 0; layer < 2; ++layer) {
    transform3_kernel<<<tiles, 32, 0, stream>>>(x, pack, b_rg, h01, root, N);
    edge_kernel<<<egrid, 256, 0, stream>>>(eidx, etyp, h01, sums, cnt, N, E,
                                           layer == 0 ? 1 : 0);
    combine_kernel<<<(N * 32 + 255) / 256, 256, 0, stream>>>(sums, cnt, root, x, N);
    if (layer == 0) {                    // re-zero sums for layer 2; keep cnt
      const int n = 64 * N;
      zero_kernel<<<(n + 255) / 256, 256, 0, stream>>>(sums, n);
    }
  }

  head_kernel<<<tiles, 32, 0, stream>>>(x, pack, b_o1, b_o2, (float*)d_out);
}
